// DeformableConv2d_4260607558015
// MI455X (gfx1250) — matrix operand order repaired and checked by reading, not compiled, not run on hardware
//
#include <hip/hip_runtime.h>
#include <hip/hip_bf16.h>

typedef __attribute__((ext_vector_type(16))) _Float16 v16h;
typedef __attribute__((ext_vector_type(8)))  float    v8f;

#define Bn 4
#define Cn 64
#define On 64
#define Dn 2
#define Hn 128
#define Wn 128
#define HW (Hn*Wn)
#define K2 9
#define CK (Cn*K2)          // 576
#define KSTEPS (CK/32)      // 18
#define AW_ELEMS (4*KSTEPS*32*16)   // 36864 f16 = 73728 bytes
#define NPIX 64             // pixels per block in deform_gemm (4 N-tiles)

// ---------------------------------------------------------------------------
// Kernel W: repack w0 (O,C,3,3) fp32 -> f16 WMMA A-fragments.
// Fragment storage: Aw[((m_tile*18 + ks)*32 + lane)*16 + h]
// ISA 16-bit A 16x32 layout: M = lane%16, K = (h>=8)*16 + (lane>=16)*8 + h%8
// ---------------------------------------------------------------------------
__global__ __launch_bounds__(256) void repack_w(const float* __restrict__ w0,
                                                _Float16* __restrict__ Aw) {
    int idx = blockIdx.x * 256 + threadIdx.x;
    if (idx >= AW_ELEMS) return;
    int h    = idx & 15;
    int lane = (idx >> 4) & 31;
    int ks   = (idx >> 9) % KSTEPS;
    int mt   = idx / (KSTEPS * 512);
    int M = lane & 15;
    int K = ((h >> 3) << 4) + ((lane >> 4) << 3) + (h & 7);
    int k = ks * 32 + K;             // k = c*9 + tap  (w0 flattened (O, C*9))
    int m = mt * 16 + M;
    Aw[idx] = (_Float16)w0[m * CK + k];
}

// ---------------------------------------------------------------------------
// Kernel A: conv3d (3x3x3, pad 1) producing 27 channels:
//   ch 0..17  -> raw offsets (bias added)
//   ch 18..26 -> 2*sigmoid(conv + bias)  (modulation mask)
// Output layout: om[(b*27 + ch)*Dn*HW + d*HW + pix]
// ---------------------------------------------------------------------------
__global__ __launch_bounds__(256) void convmask(const float* __restrict__ x,
                                                const float* __restrict__ ow,
                                                const float* __restrict__ ob,
                                                const float* __restrict__ mw,
                                                const float* __restrict__ mb,
                                                float* __restrict__ om) {
    __shared__ float wsm[Cn * 27];
    int tid = threadIdx.x;
    int ch = blockIdx.y >> 1;
    int d  = blockIdx.y & 1;
    int b  = blockIdx.z;
    const float* wsrc = (ch < 18) ? (ow + ch * Cn * 27) : (mw + (ch - 18) * Cn * 27);
    float bias = (ch < 18) ? ob[ch] : mb[ch - 18];
    for (int i = tid; i < Cn * 27; i += 256) wsm[i] = wsrc[i];
    __syncthreads();

    int pix = blockIdx.x * 256 + tid;
    int h = pix >> 7, w = pix & (Wn - 1);
    float acc = 0.f;
    for (int ic = 0; ic < Cn; ++ic) {
        const float* xc = x + ((size_t)(b * Cn + ic) * Dn) * HW;
        const float* wk = wsm + ic * 27;
        for (int kd = 0; kd < 3; ++kd) {
            int di = d + kd - 1;
            if (di < 0 || di >= Dn) continue;
            const float* xd = xc + (size_t)di * HW;
            for (int kh = 0; kh < 3; ++kh) {
                int yi = h + kh - 1;
                if (yi < 0 || yi >= Hn) continue;
                const float* xr = xd + yi * Wn;
                for (int kw = 0; kw < 3; ++kw) {
                    int xi = w + kw - 1;
                    if (xi < 0 || xi >= Wn) continue;
                    acc += wk[kd * 9 + kh * 3 + kw] * xr[xi];
                }
            }
        }
    }
    float t = acc + bias;
    if (ch >= 18) t = 2.f / (1.f + __expf(-t));
    om[((size_t)(b * 27 + ch) * Dn + d) * HW + pix] = t;
}

// ---------------------------------------------------------------------------
// Kernel B: fused deformable sampling + modulated GEMM via f16 WMMA.
// One block = one (b,d) and 64 consecutive pixels (4 N-tiles).
//   Stage 1: build B-fragments (patches * mask) in LDS, f16.
//            B layout (16-bit, 32x16): n = lane%16, K = (half>=8)*16 + (lane>=16)*8 + half%8
//   Stage 2: 8 waves = 4 M-tiles x 2 N-pairs; each wave keeps TWO f32
//            accumulators so one A-fragment load feeds two WMMAs
//            (18 k-steps -> 36 x v_wmma_f32_16x16x32_f16 per wave).
// ---------------------------------------------------------------------------
__global__ __launch_bounds__(256) void deform_gemm(const float* __restrict__ x,
                                                   const float* __restrict__ om,
                                                   const _Float16* __restrict__ Aw,
                                                   float* __restrict__ out) {
    __shared__ __align__(32) _Float16 Bsm[4][KSTEPS][32 * 16];  // 72 KB
    int tid = threadIdx.x;
    int d = blockIdx.y, b = blockIdx.z;
    int pixBase = blockIdx.x * NPIX;

    // ---- Stage 1: bilinear sample + modulate -> LDS B fragments ----
    const size_t cs = (size_t)Dn * HW;          // channel stride in om / x
    for (int pair = tid; pair < NPIX * K2; pair += 256) {
        int p = pair / K2, tap = pair - p * K2;
        int pix = pixBase + p;
        int h = pix >> 7, w = pix & (Wn - 1);
        size_t obase = ((size_t)(b * 27) * Dn + d) * HW + pix;
        float dy = om[obase + (size_t)(tap * 2 + 0) * cs];
        float dx = om[obase + (size_t)(tap * 2 + 1) * cs];
        float mv = om[obase + (size_t)(18 + tap) * cs];

        float py = (float)(h - 1 + tap / 3) + dy;
        float px = (float)(w - 1 + tap % 3) + dx;
        float y0f = floorf(py), x0f = floorf(px);
        float ly = py - y0f, lx = px - x0f;
        int y0 = (int)y0f, x0 = (int)x0f;
        int y1 = y0 + 1, x1 = x0 + 1;
        bool vy0 = (y0 >= 0) && (y0 < Hn);
        bool vy1 = (y1 >= 0) && (y1 < Hn);
        bool vx0 = (x0 >= 0) && (x0 < Wn);
        bool vx1 = (x1 >= 0) && (x1 < Wn);
        float w00 = (1.f - ly) * (1.f - lx) * ((vy0 && vx0) ? 1.f : 0.f);
        float w01 = (1.f - ly) * lx         * ((vy0 && vx1) ? 1.f : 0.f);
        float w10 = ly * (1.f - lx)         * ((vy1 && vx0) ? 1.f : 0.f);
        float w11 = ly * lx                 * ((vy1 && vx1) ? 1.f : 0.f);
        int cy0 = min(max(y0, 0), Hn - 1), cy1 = min(max(y1, 0), Hn - 1);
        int cx0 = min(max(x0, 0), Wn - 1), cx1 = min(max(x1, 0), Wn - 1);
        int i00 = cy0 * Wn + cx0, i01 = cy0 * Wn + cx1;
        int i10 = cy1 * Wn + cx0, i11 = cy1 * Wn + cx1;

        const float* xb = x + ((size_t)b * Cn * Dn + d) * HW;
        int nt = p >> 4, n = p & 15;
        for (int c = 0; c < Cn; ++c) {
            const float* xc = xb + (size_t)c * cs;
            float v = w00 * xc[i00] + w01 * xc[i01] + w10 * xc[i10] + w11 * xc[i11];
            v *= mv;
            int k = c * K2 + tap;
            int ks = k >> 5, K = k & 31;
            int lane = (((K >> 3) & 1) << 4) | n;
            Bsm[nt][ks][lane * 16 + ((K & 7) | ((K >> 4) << 3))] = (_Float16)v;
        }
    }
    __syncthreads();

    // ---- Stage 2: WMMA contraction over K = 576, 2 N-tiles per wave ----
    int wid = tid >> 5, lane = tid & 31;
    int m_tile = wid & 3;
    int nt0 = (wid >> 2) * 2;        // N-tiles nt0 and nt0+1
    v8f acc0 = {0.f, 0.f, 0.f, 0.f, 0.f, 0.f, 0.f, 0.f};
    v8f acc1 = {0.f, 0.f, 0.f, 0.f, 0.f, 0.f, 0.f, 0.f};
    const v16h* Ag = (const v16h*)Aw;
    for (int ks = 0; ks < KSTEPS; ++ks) {
        v16h a   = Ag[(m_tile * KSTEPS + ks) * 32 + lane];
        v16h bf0 = *(const v16h*)&Bsm[nt0 + 0][ks][lane * 16];
        v16h bf1 = *(const v16h*)&Bsm[nt0 + 1][ks][lane * 16];
        acc0 = __builtin_amdgcn_wmma_f32_16x16x32_f16(
            false, a, false, bf0, (short)0, acc0, false, false);
        acc1 = __builtin_amdgcn_wmma_f32_16x16x32_f16(
            false, a, false, bf1, (short)0, acc1, false, false);
    }

    // ---- D layout: VGPR r: lanes 0-15 -> M=r, lanes 16-31 -> M=8+r; N=lane%16
    int n = lane & 15;
    int mrow = m_tile * 16 + ((lane >> 4) << 3);
    int pix0 = pixBase + (nt0 + 0) * 16 + n;
    int pix1 = pixBase + (nt0 + 1) * 16 + n;
#pragma unroll
    for (int r = 0; r < 8; ++r) {
        int o = mrow + r;
        size_t ch_base = ((size_t)(b * On + o) * Dn + d) * HW;
        out[ch_base + pix0] = acc0[r];
        out[ch_base + pix1] = acc1[r];
    }
}

// ---------------------------------------------------------------------------
extern "C" void kernel_launch(void* const* d_in, const int* in_sizes, int n_in,
                              void* d_out, int out_size, void* d_ws, size_t ws_size,
                              hipStream_t stream) {
    const float* x  = (const float*)d_in[0];
    const float* ow = (const float*)d_in[1];
    const float* ob = (const float*)d_in[2];
    const float* mw = (const float*)d_in[3];
    const float* mb = (const float*)d_in[4];
    const float* w0 = (const float*)d_in[5];
    float* out = (float*)d_out;

    char* ws = (char*)d_ws;
    _Float16* Aw = (_Float16*)ws;                               // 73,728 B
    float* om = (float*)(ws + AW_ELEMS * sizeof(_Float16));     // 14,155,776 B

    // 1) repack weights into WMMA A fragments (f16)
    repack_w<<<AW_ELEMS / 256, 256, 0, stream>>>(w0, Aw);

    // 2) offset/mask conv3d + fused 2*sigmoid
    convmask<<<dim3(HW / 256, 27 * Dn, Bn), 256, 0, stream>>>(x, ow, ob, mw, mb, om);

    // 3) fused deformable sampling + WMMA GEMM (64 pixels / block)
    deform_gemm<<<dim3(HW / NPIX, Dn, Bn), 256, 0, stream>>>(x, om, Aw, out);
}
